// Attention_66795331388102
// MI455X (gfx1250) — matrix operand order repaired and checked by reading, not compiled, not run on hardware
//
#include <hip/hip_runtime.h>
#include <hip/hip_bf16.h>

// ---------------- problem constants ----------------
#define QN      8192
#define NW      5
#define DIM     640
#define INNER   512
#define HEADS   8
#define DH      64

#define ROWS_Q  (QN * NW)            // 40960
#define ROWS_ALL (ROWS_Q + 2 * QN)   // 57344
#define ROW_K_OFF ROWS_Q             // 40960
#define ROW_V_OFF (ROWS_Q + QN)      // 49152

// ---------------- types for WMMA / TDM -------------------
typedef __attribute__((ext_vector_type(16))) __bf16 bf16x16;
typedef __attribute__((ext_vector_type(8)))  float  f32x8;
typedef __attribute__((ext_vector_type(4)))  unsigned int u32x4;
typedef __attribute__((ext_vector_type(8)))  unsigned int u32x8;

union FragAB {            // 32 bytes: two 16B LDS loads -> one WMMA operand
    uint4   u[2];
    bf16x16 v;
};

// ---------------- small helpers --------------------
__device__ __forceinline__ unsigned short f2bf(float f) {
    unsigned u = __float_as_uint(f);
    unsigned r = u + 0x7FFFu + ((u >> 16) & 1u);   // round-to-nearest-even
    return (unsigned short)(r >> 16);
}
__device__ __forceinline__ float bf2f(unsigned short s) {
    return __uint_as_float(((unsigned)s) << 16);
}
__device__ __forceinline__ float wave_sum(float v) {
    #pragma unroll
    for (int m = 16; m > 0; m >>= 1) v += __shfl_xor(v, m, 32);
    return v;
}

// Flat shared-aperture address -> wave-relative LDS byte address (addr[31:0]).
__device__ __forceinline__ unsigned lds_addr_of(const void* p) {
    return (unsigned)(uintptr_t)p;
}

// ASYNCcnt-tracked 16B global->LDS copy (no VGPR staging).
__device__ __forceinline__ void async_copy_b128(const void* gsrc, void* ldst) {
    unsigned           ld = lds_addr_of(ldst);
    unsigned long long ga = (unsigned long long)(uintptr_t)gsrc;
    asm volatile("global_load_async_to_lds_b128 %0, %1, off"
                 :: "v"(ld), "v"(ga) : "memory");
}
__device__ __forceinline__ void wait_async0() {
    asm volatile("s_wait_asynccnt 0x0" ::: "memory");
}

// TDM: DMA a 64x64 bf16 tile (row stride = row_stride_elems) from global
// memory into LDS, inserting 16B padding after every 128B row so LDS rows
// land at stride 144B (matching sB[r][72]).  One instruction per wave,
// tracked with TENSORcnt.  D# per CDNA5 ISA.
__device__ __forceinline__ void tdm_load_tile64(
    const unsigned short* gbase, unsigned lds_byte_addr, int row_stride_elems)
{
    unsigned long long ga = (unsigned long long)(uintptr_t)gbase;
    u32x4 g0;
    g0[0] = 1u;                                          // count=1 (valid user D#)
    g0[1] = lds_byte_addr;                               // lds_addr [63:32]
    g0[2] = (unsigned)ga;                                // global_addr[31:0]
    g0[3] = (unsigned)((ga >> 32) & 0x1FFFFFFu)          // global_addr[56:32]
          | (2u << 30);                                  // type=2 ("image")
    u32x8 g1;
    g1[0] = (1u << 16)                                   // data_size = 2 bytes
          | (1u << 20)                                   // pad_enable
          | (4u << 22)                                   // pad_interval: 32 DW = 128B
          | (3u << 25);                                  // pad_amount:    4 DW = 16B
    g1[1] = 0u;                                          // no atomic barrier; dim0 lo
    g1[2] = 0x10u;                                       // tensor_dim0 = 1M (no OOB)
    g1[3] = 0x10u | (64u << 16);                         // tensor_dim1 = 1M; tile_dim0=64
    g1[4] = 64u;                                         // tile_dim1 = 64; tile_dim2 = 0
    g1[5] = (unsigned)row_stride_elems;                  // tensor_dim0_stride
    g1[6] = 0u;
    g1[7] = 0u;
    u32x4 gz = {0u, 0u, 0u, 0u};                         // groups 2/3: unused dims
    asm volatile("tensor_load_to_lds %0, %1, %2, %3"
                 :: "s"(g0), "s"(g1), "s"(gz), "s"(gz) : "memory");
}

// ======================================================================
// Kernel 0: W_in (640x512) -> WinT bf16 (512x640) ; W_out (512x640) ->
//           WoutT bf16 (640x512).
// ======================================================================
__global__ __launch_bounds__(256) void convert_weights_kernel(
    const float* __restrict__ Win, const float* __restrict__ Wout,
    unsigned short* __restrict__ WinT, unsigned short* __restrict__ WoutT)
{
    int i = blockIdx.x * 256 + threadIdx.x;          // 0 .. 655359
    if (i < DIM * INNER) {                           // WinT[n][k] = Win[k][n]
        int n = i / DIM, kd = i - n * DIM;
        WinT[i] = f2bf(Win[(size_t)kd * INNER + n]);
    } else {
        int j = i - DIM * INNER;                     // WoutT[c][k] = Wout[k][c]
        int c = j / INNER, kk = j - c * INNER;
        WoutT[j] = f2bf(Wout[(size_t)kk * DIM + c]);
    }
}

// ======================================================================
// Kernel 1: fused LayerNorm over q|k|v rows -> bf16 xnorm (57344 x 640).
// ======================================================================
__global__ __launch_bounds__(256) void ln_kernel(
    const float* __restrict__ q, const float* __restrict__ k,
    const float* __restrict__ v, const float* __restrict__ g,
    const float* __restrict__ b, unsigned short* __restrict__ xnorm)
{
    int wave = threadIdx.x >> 5, lane = threadIdx.x & 31;
    int row  = blockIdx.x * 8 + wave;

    const float* src;
    if (row < ROW_K_OFF)      src = q + (size_t)row * DIM;
    else if (row < ROW_V_OFF) src = k + (size_t)(row - ROW_K_OFF) * DIM;
    else                      src = v + (size_t)(row - ROW_V_OFF) * DIM;

    float x[20], s = 0.f, ss = 0.f;
    #pragma unroll
    for (int i = 0; i < 20; ++i) {
        x[i] = src[lane + 32 * i];
        s += x[i];  ss += x[i] * x[i];
    }
    s  = wave_sum(s);
    ss = wave_sum(ss);
    float mean = s * (1.f / DIM);
    float var  = ss * (1.f / DIM) - mean * mean;     // biased, like jnp.var
    float rstd = rsqrtf(var + 1e-5f);

    unsigned short* dst = xnorm + (size_t)row * DIM;
    #pragma unroll
    for (int i = 0; i < 20; ++i) {
        int c = lane + 32 * i;
        dst[c] = f2bf((x[i] - mean) * rstd * g[c] + b[c]);
    }
}

// ======================================================================
// Tiled bf16 GEMM:  C[M,N] = A[M,K] (bf16, row major) * BT[N,K]^T (+bias)
// Block tile 128x64, K-chunk 64, double-buffered LDS.  A tile filled with
// global_load_async_to_lds_b128 (ASYNCcnt); B tile DMA'd by the Tensor
// Data Mover (TENSORcnt, wave 0 issues).  8 waves = 4x2 grid, each wave a
// 32x32 patch = 2x2 v_wmma_f32_16x16x32_bf16 tiles.  LDS rows padded to
// 72 bf16 (144B = 9*16B): conflict-free fragment loads.
// ======================================================================
template<bool OUT_BF16, bool HAS_BIAS>
__global__ __launch_bounds__(256) void gemm_bf16_kernel(
    const unsigned short* __restrict__ A, int lda,
    const unsigned short* __restrict__ BT,
    void* __restrict__ Cout, int ldc,
    const float* __restrict__ bias, int K)
{
    __shared__ unsigned short sA[2][128][72];   // 2 x 18432 B
    __shared__ unsigned short sB[2][64][72];    // 2 x  9216 B

    const int row0 = blockIdx.x * 128;
    const int col0 = blockIdx.y * 64;
    const int wave = threadIdx.x >> 5, lane = threadIdx.x & 31;
    const int wm = wave >> 1, wn = wave & 1;         // 4x2 wave grid
    const int half = lane >> 4, l15 = lane & 15;

    // issue all transfers for K-chunk kc into buffer bf
    auto issue_chunk = [&](int kc, int bf) {
        #pragma unroll
        for (int c = 0; c < 4; ++c) {
            int ci = threadIdx.x + 256 * c;
            int r = ci >> 3, kk = (ci & 7) << 3;
            async_copy_b128(&A[(size_t)(row0 + r) * lda + kc + kk],
                            &sA[bf][r][kk]);
        }
        if (wave == 0)
            tdm_load_tile64(&BT[(size_t)col0 * K + kc],
                            lds_addr_of(&sB[bf][0][0]), K);
    };
    auto drain = [&]() {                 // my wave's transfers done -> barrier
        wait_async0();
        if (wave == 0) __builtin_amdgcn_s_wait_tensorcnt(0);
        __syncthreads();
    };

    f32x8 acc[2][2] = {};

    issue_chunk(0, 0);
    drain();

    int buf = 0;
    for (int kc = 0; kc < K; kc += 64) {
        if (kc + 64 < K) issue_chunk(kc + 64, buf ^ 1);   // prefetch next chunk

        #pragma unroll
        for (int ks = 0; ks < 64; ks += 32) {
            FragAB a[2], b[2];
            #pragma unroll
            for (int i = 0; i < 2; ++i) {
                // A 16x32 bf16 layout: lanes 0-15 K{0..7,16..23},
                //                      lanes 16-31 K{8..15,24..31}
                int r = wm * 32 + i * 16 + l15;
                a[i].u[0] = *(const uint4*)&sA[buf][r][ks + half * 8];
                a[i].u[1] = *(const uint4*)&sA[buf][r][ks + 16 + half * 8];
            }
            #pragma unroll
            for (int j = 0; j < 2; ++j) {
                // B 32x16: lanes 0-15 K{0..7,16..23}, lanes 16-31 K{8..15,24..31} of column n
                int n = wn * 32 + j * 16 + l15;
                b[j].u[0] = *(const uint4*)&sB[buf][n][ks + half * 8];
                b[j].u[1] = *(const uint4*)&sB[buf][n][ks + 16 + half * 8];
            }
            #pragma unroll
            for (int i = 0; i < 2; ++i)
                #pragma unroll
                for (int j = 0; j < 2; ++j)
                    acc[i][j] = __builtin_amdgcn_wmma_f32_16x16x32_bf16(
                        false, a[i].v, false, b[j].v,
                        (short)0, acc[i][j], false, false);
        }
        drain();          // next chunk ready + everyone done reading this one
        buf ^= 1;
    }

    // ---- store C (layout: lanes 0-15 = M 0..7, lanes 16-31 = M 8..15) ----
    #pragma unroll
    for (int i = 0; i < 2; ++i)
        #pragma unroll
        for (int j = 0; j < 2; ++j)
            #pragma unroll
            for (int e = 0; e < 8; ++e) {
                int rr = row0 + wm * 32 + i * 16 + half * 8 + e;
                int cc = col0 + wn * 32 + j * 16 + l15;
                float val = acc[i][j][e];
                if (HAS_BIAS) val += bias[cc];
                if (OUT_BF16)
                    ((unsigned short*)Cout)[(size_t)rr * ldc + cc] = f2bf(val);
                else
                    ((float*)Cout)[(size_t)rr * ldc + cc] = val;
            }
}

// ======================================================================
// Kernel 3: attention core.  Block per q, wave per head; lane l owns
// d-elements {l, l+32}.  Wave32 shuffle reductions only.
// ======================================================================
__global__ __launch_bounds__(256) void attn_kernel(
    const unsigned short* __restrict__ F, unsigned short* __restrict__ A2,
    const float* __restrict__ vscale, const float* __restrict__ cscale)
{
    const int qi = blockIdx.x;
    const int h = threadIdx.x >> 5, lane = threadIdx.x & 31;
    const int col = h * DH + lane;

    const size_t bk = ((size_t)(ROW_K_OFF + qi)) * INNER + col;
    const size_t bv = ((size_t)(ROW_V_OFF + qi)) * INNER + col;
    float fk0 = bf2f(F[bk]), fk1 = bf2f(F[bk + 32]);
    float fv0 = bf2f(F[bv]), fv1 = bf2f(F[bv + 32]);

    float sk  = wave_sum(fk0 + fk1);
    float skk = wave_sum(fk0 * fk0 + fk1 * fk1);
    float mean_k = sk * (1.f / DH);
    float var_k  = skk * (1.f / DH) - mean_k * mean_k;
    float nk = sqrtf(skk);

    const float vs = vscale[0], cs = cscale[0];
    float cosv[NW], vw[NW], sig[NW];
    float fv_scale0[NW], fv_scale1[NW];
    float vwsum = 0.f;
    #pragma unroll
    for (int n = 0; n < NW; ++n) {
        size_t bq = ((size_t)(qi * NW + n)) * INNER + col;
        float f0 = bf2f(F[bq]), f1 = bf2f(F[bq + 32]);
        float sq  = wave_sum(f0 + f1);
        float sqq = wave_sum(f0 * f0 + f1 * f1);
        float sd  = wave_sum(f0 * fk0 + f1 * fk1);
        float mq   = sq * (1.f / DH);
        float varq = sqq * (1.f / DH) - mq * mq;
        float nq   = sqrtf(sqq);
        cosv[n] = sd / (nq * nk);
        vw[n] = 1.f / (fabsf(var_k - varq) + 1e-6f);
        vwsum += vw[n];
        float cov = (sd - (float)DH * mq * mean_k) * (1.f / ((float)DH + 1e-6f));
        sig[n] = 1.f / (1.f + __expf(-cov));
        fv_scale0[n] = fv0;  fv_scale1[n] = fv1;
    }
    float inv = 1.f / (vwsum + 1e-6f);
    #pragma unroll
    for (int n = 0; n < NW; ++n) {
        float d = cosv[n] + vs * vw[n] * inv + cs * sig[n];
        size_t bo = ((size_t)(qi * NW + n)) * INNER + col;
        A2[bo]      = f2bf(d * fv_scale0[n]);
        A2[bo + 32] = f2bf(d * fv_scale1[n]);
    }
}

// ======================================================================
// launch
// ======================================================================
extern "C" void kernel_launch(void* const* d_in, const int* in_sizes, int n_in,
                              void* d_out, int out_size, void* d_ws, size_t ws_size,
                              hipStream_t stream)
{
    const float* q     = (const float*)d_in[0];
    const float* k     = (const float*)d_in[1];
    const float* v     = (const float*)d_in[2];
    const float* ln_g  = (const float*)d_in[3];
    const float* ln_b  = (const float*)d_in[4];
    const float* W_in  = (const float*)d_in[5];
    const float* W_out = (const float*)d_in[6];
    const float* b_out = (const float*)d_in[7];
    const float* vsc   = (const float*)d_in[8];
    const float* csc   = (const float*)d_in[9];
    float* out = (float*)d_out;

    // workspace layout (bytes), all 256B-aligned:
    //   [0,       655360)  WinT  bf16 (512 x 640)
    //   [655360, 1310720)  WoutT bf16 (640 x 512)
    //   [1310720, +73.4MB) xnorm bf16 (57344 x 640)  -- reused as A2 (40960x512)
    //   [74711040,+58.7MB) F     bf16 (57344 x 512)
    char* ws = (char*)d_ws;
    unsigned short* WinT  = (unsigned short*)(ws);
    unsigned short* WoutT = (unsigned short*)(ws + 655360);
    unsigned short* xnorm = (unsigned short*)(ws + 1310720);
    unsigned short* A2    = xnorm;                 // alias: xnorm dead after GEMM1
    unsigned short* F     = (unsigned short*)(ws + 74711040);

    convert_weights_kernel<<<2560, 256, 0, stream>>>(W_in, W_out, WinT, WoutT);

    ln_kernel<<<ROWS_ALL / 8, 256, 0, stream>>>(q, k, v, ln_g, ln_b, xnorm);

    // GEMM1: F(57344x512) = xnorm(57344x640) @ W_in   -> bf16, no bias
    gemm_bf16_kernel<true, false>
        <<<dim3(ROWS_ALL / 128, INNER / 64), 256, 0, stream>>>(
            xnorm, DIM, WinT, F, INNER, nullptr, DIM);

    attn_kernel<<<QN, 256, 0, stream>>>(F, A2, vsc, csc);

    // GEMM2: out(40960x640) = A2(40960x512) @ W_out + b_out -> f32
    gemm_bf16_kernel<false, true>
        <<<dim3(ROWS_Q / 128, DIM / 64), 256, 0, stream>>>(
            A2, INNER, WoutT, out, DIM, b_out, INNER);
}
